// LConvolutionModule_51891794870779
// MI455X (gfx1250) — matrix operand order repaired and checked by reading, not compiled, not run on hardware
//
#include <hip/hip_runtime.h>

// ---------------- problem constants ----------------
#define Bc   16
#define Tc   2048
#define Cc   512
#define Hc   4
#define Kc   31
#define TWOC 1024
#define BT   (Bc * Tc)          // 32768

typedef __attribute__((ext_vector_type(16))) __bf16 v16bf;
typedef __attribute__((ext_vector_type(8)))  float  v8f;

union BF16x16 { v16bf v; uint4 q[2]; };

__device__ __forceinline__ unsigned short f2bf(float f) {
    unsigned u = __float_as_uint(f);
    unsigned r = u + 0x7FFFu + ((u >> 16) & 1u);   // round-to-nearest-even
    return (unsigned short)(r >> 16);
}
__device__ __forceinline__ float bf2f(unsigned hu) {
    return __uint_as_float(hu << 16);
}

// 16B global -> LDS async copy (per-lane addresses), tracked by ASYNCcnt
__device__ __forceinline__ void async_cp16(unsigned lds_off, const void* gaddr) {
    asm volatile("global_load_async_to_lds_b128 %0, %1, off"
                 :: "v"(lds_off), "v"(gaddr) : "memory");
}
__device__ __forceinline__ void wait_async0() {
    asm volatile("s_wait_asynccnt 0x0" ::: "memory");
}
__device__ __forceinline__ unsigned lds_addr(const void* p) {
    return (unsigned)(unsigned long long)p;   // low 32 bits of generic ptr = LDS offset
}

// ---------------- kernel 0: weight conversion + kernel softmax ----------------
__global__ void prep_kernel(const float* __restrict__ w1, const float* __restrict__ w2,
                            const float* __restrict__ lw,
                            unsigned short* __restrict__ w1b, unsigned short* __restrict__ w2b,
                            float* __restrict__ wsm4) {
    const int n1 = TWOC * Cc;           // 524288
    const int n2 = Cc * Cc;             // 262144
    int idx = blockIdx.x * blockDim.x + threadIdx.x;
    if (idx < n1)            w1b[idx]       = f2bf(w1[idx]);
    else if (idx < n1 + n2)  w2b[idx - n1]  = f2bf(w2[idx - n1]);

    if (blockIdx.x == 0 && threadIdx.x < Hc) {
        int h = threadIdx.x;
        float mx = -1e30f;
        for (int j = 0; j < Kc; ++j) mx = fmaxf(mx, lw[h * Kc + j]);
        float s = 0.f;
        for (int j = 0; j < Kc; ++j) s += __expf(lw[h * Kc + j] - mx);
        float inv = 1.f / s;
        for (int j = 0; j < Kc; ++j)
            wsm4[j * Hc + h] = __expf(lw[h * Kc + j] - mx) * inv;   // layout [K][H]
    }
}

// ---------------- kernel 1: GEMM1 (x @ w1^T + b1) + GLU -> yg (bf16) ----------------
// tile: 64 rows (BT) x 64 cols (of yg, i.e. 128 cols of y), K-step 32
__global__ void __launch_bounds__(256)
gemm1_glu_kernel(const float* __restrict__ x, const unsigned short* __restrict__ w1b,
                 const float* __restrict__ b1, unsigned short* __restrict__ yg) {
    __shared__ unsigned short As[64 * 40];    // row stride 40 elems (80B) -> conflict-free b128
    __shared__ unsigned short Bs[128 * 40];   // rows 0..63: a-half d, rows 64..127: g-half d+512

    const int tid  = threadIdx.x;
    const int lane = tid & 31;
    const int wave = tid >> 5;
    const int rm   = wave & 3;     // 16-row strip
    const int cp   = wave >> 2;    // 0/1: which pair of 16-col tiles

    const int bt0 = blockIdx.x * 64;
    const int c0  = blockIdx.y * 64;

    v8f acc_a[2] = {}; v8f acc_g[2] = {};

    const int m  = lane & 15;
    const int hi = lane >> 4;

    // B-stage async addressing (2 chunks of 16B per thread per step)
    int brow0 = tid >> 2,         bpart0 = tid & 3;
    int brow1 = (tid + 256) >> 2, bpart1 = (tid + 256) & 3;
    int d0 = (brow0 < 64) ? (c0 + brow0) : (Cc + c0 + brow0 - 64);
    int d1 = (brow1 < 64) ? (c0 + brow1) : (Cc + c0 + brow1 - 64);
    const unsigned short* gB0 = w1b + (size_t)d0 * Cc + bpart0 * 8;
    const unsigned short* gB1 = w1b + (size_t)d1 * Cc + bpart1 * 8;
    const unsigned ldsB0 = lds_addr(&Bs[brow0 * 40 + bpart0 * 8]);
    const unsigned ldsB1 = lds_addr(&Bs[brow1 * 40 + bpart1 * 8]);

    for (int kk0 = 0; kk0 < Cc; kk0 += 32) {
        // ---- stage B via async copy (pure bf16 copy, bypasses VGPRs) ----
        async_cp16(ldsB0, gB0 + kk0);
        async_cp16(ldsB1, gB1 + kk0);

        // ---- stage A: 64x32 fp32 -> bf16 LDS (needs conversion, VGPR path) ----
        #pragma unroll
        for (int t = 0; t < 2; ++t) {
            int idx = tid + t * 256;            // 512 float4 tasks
            int row = idx >> 3, c4 = idx & 7;
            float4 xv = *(const float4*)(x + (size_t)(bt0 + row) * Cc + kk0 + c4 * 4);
            uint2 p;
            p.x = (unsigned)f2bf(xv.x) | ((unsigned)f2bf(xv.y) << 16);
            p.y = (unsigned)f2bf(xv.z) | ((unsigned)f2bf(xv.w) << 16);
            *(uint2*)&As[row * 40 + c4 * 4] = p;
        }
        wait_async0();
        __syncthreads();

        // ---- A fragment (16x32 bf16 layout) ----
        BF16x16 af;
        const unsigned short* ap = &As[(rm * 16 + m) * 40];
        af.q[0] = *(const uint4*)(ap + hi * 8);          // K = kb .. kb+7
        af.q[1] = *(const uint4*)(ap + 16 + hi * 8);     // K = kb+16 .. kb+23

        // ---- B fragments + WMMA ----
        #pragma unroll
        for (int half = 0; half < 2; ++half) {
            #pragma unroll
            for (int j = 0; j < 2; ++j) {
                int brow = half * 64 + cp * 32 + j * 16 + m;
                BF16x16 bf;
                const unsigned short* bp = &Bs[brow * 40 + hi * 8];
                bf.q[0] = *(const uint4*)(bp);
                bf.q[1] = *(const uint4*)(bp + 16);
                v8f acc = half ? acc_g[j] : acc_a[j];
                acc = __builtin_amdgcn_wmma_f32_16x16x32_bf16(
                        false, af.v, false, bf.v, (short)0, acc, false, false);
                if (half) acc_g[j] = acc; else acc_a[j] = acc;
            }
        }
        __syncthreads();
    }

    // ---- epilogue: bias + GLU, store bf16 ----
    #pragma unroll
    for (int j = 0; j < 2; ++j) {
        int d  = c0 + (cp * 2 + j) * 16 + m;
        float ba = b1[d];
        float bg = b1[d + Cc];
        #pragma unroll
        for (int r = 0; r < 8; ++r) {
            int mrow = hi * 8 + r;
            float a = acc_a[j][r] + ba;
            float g = acc_g[j][r] + bg;
            float v = a * (1.0f / (1.0f + __expf(-g)));
            yg[(size_t)(bt0 + rm * 16 + mrow) * Cc + d] = f2bf(v);
        }
    }
}

// ---------------- kernel 2: depthwise LightConv along T + lconv_bias ----------------
// block: 64 T-rows x 64 channels, halo 15 each side staged in LDS
__global__ void __launch_bounds__(256)
lconv_kernel(const unsigned short* __restrict__ yg, const float* __restrict__ wsm4,
             const float* __restrict__ lbias, unsigned short* __restrict__ convb) {
    __shared__ unsigned short Ys[94][72];   // 144B row stride (16B multiple)
    __shared__ float Ws[Kc * Hc];           // [j][h]

    const int tid = threadIdx.x;
    const int t0  = blockIdx.x * 64;
    const int b   = blockIdx.y;
    const int c0  = blockIdx.z * 64;

    if (tid < Kc * Hc) Ws[tid] = wsm4[tid];

    for (int i = tid; i < 94 * 8; i += 256) {       // 16B chunks
        int row = i >> 3, part = i & 7;
        int t = t0 - 15 + row;
        uint4 v = make_uint4(0u, 0u, 0u, 0u);
        if (t >= 0 && t < Tc)
            v = *(const uint4*)(yg + ((size_t)(b * Tc + t)) * Cc + c0 + part * 8);
        *(uint4*)&Ys[row][part * 8] = v;
    }
    __syncthreads();

    for (int i = tid; i < 64 * 8; i += 256) {
        int r = i >> 3, part = i & 7;
        int c = c0 + part * 8;
        float4 bl0 = *(const float4*)(lbias + c);
        float4 bl1 = *(const float4*)(lbias + c + 4);
        float acc[8] = { bl0.x, bl0.y, bl0.z, bl0.w, bl1.x, bl1.y, bl1.z, bl1.w };
        #pragma unroll
        for (int j = 0; j < Kc; ++j) {
            float4 w4 = *(const float4*)&Ws[j * 4];
            uint4 yv = *(const uint4*)&Ys[r + j][part * 8];
            acc[0] += w4.x * bf2f(yv.x & 0xFFFFu);
            acc[1] += w4.y * bf2f(yv.x >> 16);
            acc[2] += w4.z * bf2f(yv.y & 0xFFFFu);
            acc[3] += w4.w * bf2f(yv.y >> 16);
            acc[4] += w4.x * bf2f(yv.z & 0xFFFFu);
            acc[5] += w4.y * bf2f(yv.z >> 16);
            acc[6] += w4.z * bf2f(yv.w & 0xFFFFu);
            acc[7] += w4.w * bf2f(yv.w >> 16);
        }
        uint4 o;
        o.x = (unsigned)f2bf(acc[0]) | ((unsigned)f2bf(acc[1]) << 16);
        o.y = (unsigned)f2bf(acc[2]) | ((unsigned)f2bf(acc[3]) << 16);
        o.z = (unsigned)f2bf(acc[4]) | ((unsigned)f2bf(acc[5]) << 16);
        o.w = (unsigned)f2bf(acc[6]) | ((unsigned)f2bf(acc[7]) << 16);
        *(uint4*)(convb + ((size_t)(b * Tc + t0 + r)) * Cc + c) = o;
    }
}

// ---------------- kernel 3: GEMM2 (conv @ w2^T + b2) -> out fp32 ----------------
// Double-buffered LDS, both operands staged with async-to-LDS (ASYNCcnt),
// next tile's copy overlaps current tile's WMMAs.
__global__ void __launch_bounds__(256)
gemm2_kernel(const unsigned short* __restrict__ convb, const unsigned short* __restrict__ w2b,
             const float* __restrict__ b2, float* __restrict__ out) {
    __shared__ unsigned short As[2][64 * 40];
    __shared__ unsigned short Bs[2][64 * 40];

    const int tid  = threadIdx.x;
    const int lane = tid & 31;
    const int wave = tid >> 5;
    const int rm   = wave & 3;
    const int cp   = wave >> 2;

    const int bt0 = blockIdx.x * 64;
    const int c0  = blockIdx.y * 64;

    v8f acc[2] = {};
    const int m  = lane & 15;
    const int hi = lane >> 4;

    const int srow = tid >> 2, spart = tid & 3;     // one 16B chunk per thread per buffer
    const unsigned short* gA = convb + (size_t)(bt0 + srow) * Cc + spart * 8;
    const unsigned short* gB = w2b   + (size_t)(c0  + srow) * Cc + spart * 8;
    unsigned ldsA[2], ldsB[2];
    ldsA[0] = lds_addr(&As[0][srow * 40 + spart * 8]);
    ldsA[1] = lds_addr(&As[1][srow * 40 + spart * 8]);
    ldsB[0] = lds_addr(&Bs[0][srow * 40 + spart * 8]);
    ldsB[1] = lds_addr(&Bs[1][srow * 40 + spart * 8]);

    // prologue: stage k-step 0 into buffer 0
    async_cp16(ldsA[0], gA);
    async_cp16(ldsB[0], gB);

    const int NSTEP = Cc / 32;                      // 16
    for (int s = 0; s < NSTEP; ++s) {
        const int p = s & 1;
        wait_async0();          // my buffer-p copies landed in LDS
        __syncthreads();        // everyone's landed; everyone's reads of buffer p^1 are done

        if (s + 1 < NSTEP) {    // kick off next tile into the other buffer
            async_cp16(ldsA[p ^ 1], gA + (s + 1) * 32);
            async_cp16(ldsB[p ^ 1], gB + (s + 1) * 32);
        }

        BF16x16 af;
        const unsigned short* ap = &As[p][(rm * 16 + m) * 40];
        af.q[0] = *(const uint4*)(ap + hi * 8);
        af.q[1] = *(const uint4*)(ap + 16 + hi * 8);

        #pragma unroll
        for (int j = 0; j < 2; ++j) {
            int brow = cp * 32 + j * 16 + m;
            BF16x16 bf;
            const unsigned short* bp = &Bs[p][brow * 40 + hi * 8];
            bf.q[0] = *(const uint4*)(bp);
            bf.q[1] = *(const uint4*)(bp + 16);
            acc[j] = __builtin_amdgcn_wmma_f32_16x16x32_bf16(
                        false, af.v, false, bf.v, (short)0, acc[j], false, false);
        }
    }

    #pragma unroll
    for (int j = 0; j < 2; ++j) {
        int d = c0 + (cp * 2 + j) * 16 + m;
        float bb = b2[d];
        #pragma unroll
        for (int r = 0; r < 8; ++r) {
            int mrow = hi * 8 + r;
            out[(size_t)(bt0 + rm * 16 + mrow) * Cc + d] = acc[j][r] + bb;
        }
    }
}

// ---------------- launch ----------------
extern "C" void kernel_launch(void* const* d_in, const int* in_sizes, int n_in,
                              void* d_out, int out_size, void* d_ws, size_t ws_size,
                              hipStream_t stream) {
    const float* x     = (const float*)d_in[0];
    const float* lw    = (const float*)d_in[1];
    const float* lbias = (const float*)d_in[2];
    const float* w1    = (const float*)d_in[3];
    const float* b1    = (const float*)d_in[4];
    const float* w2    = (const float*)d_in[5];
    const float* b2    = (const float*)d_in[6];

    char* ws = (char*)d_ws;
    const size_t off_w1b  = 0;                                    // 1 MiB
    const size_t off_w2b  = off_w1b + (size_t)TWOC * Cc * 2;      // +512 KiB
    const size_t off_wsm  = off_w2b + (size_t)Cc * Cc * 2;        // +4 KiB
    const size_t off_yg   = off_wsm + 4096;                       // 32 MiB
    const size_t off_conv = off_yg + (size_t)BT * Cc * 2;         // 32 MiB

    unsigned short* w1b   = (unsigned short*)(ws + off_w1b);
    unsigned short* w2b   = (unsigned short*)(ws + off_w2b);
    float*          wsm   = (float*)(ws + off_wsm);
    unsigned short* yg    = (unsigned short*)(ws + off_yg);
    unsigned short* convb = (unsigned short*)(ws + off_conv);

    prep_kernel<<<(TWOC * Cc + Cc * Cc) / 256, 256, 0, stream>>>(w1, w2, lw, w1b, w2b, wsm);
    gemm1_glu_kernel<<<dim3(BT / 64, Cc / 64), 256, 0, stream>>>(x, w1b, b1, yg);
    lconv_kernel<<<dim3(Tc / 64, Bc, Cc / 64), 256, 0, stream>>>(yg, wsm, lbias, convb);
    gemm2_kernel<<<dim3(BT / 64, Cc / 64), 256, 0, stream>>>(convb, w2b, b2, (float*)d_out);
}
